// StockElmanCell_29978871726519
// MI455X (gfx1250) — matrix operand order repaired and checked by reading, not compiled, not run on hardware
//
#include <hip/hip_runtime.h>
#include <math.h>

#define TT 512
#define BB 32
#define DD 1024

typedef __attribute__((ext_vector_type(16))) _Float16 v16h;
typedef __attribute__((ext_vector_type(8)))  _Float16 v8h;
typedef __attribute__((ext_vector_type(8)))  float    v8f;

// ---------------------------------------------------------------------------
// Pack [DD, DD] fp32 weight W (row-major, used as h @ W^T) into f16 B-matrix
// tiles for V_WMMA_F32_16X16X32_F16 (ISA 7.12.2, 16-bit B 32x16):
//   lane l holds column n = l%16; elems 0..7 -> K = 8*(l/16) + e ; elems 8..15 -> K = 16 + 8*(l/16) + (e-8).
//   B[k][n] = W[n][k].
// Packed index: o = ((ntile*(DD/32) + ktile)*32 + lane)*16 + e
// -> each lane's 16 halves (32 B) are contiguous.
// ---------------------------------------------------------------------------
__global__ __launch_bounds__(256)
void pack_weight_f16(const float* __restrict__ W, _Float16* __restrict__ Wp) {
  size_t o = (size_t)blockIdx.x * blockDim.x + threadIdx.x;
  if (o >= (size_t)DD * DD) return;
  int e    = (int)(o & 15);
  int lane = (int)((o >> 4) & 31);
  int kt   = (int)((o >> 9) & (DD / 32 - 1));
  int nt   = (int)(o >> 14);
  int n = nt * 16 + (lane & 15);
  int k = kt * 32 + 8 * (lane >> 4) + (e & 7) + 16 * (e >> 3);
  Wp[o] = (_Float16)W[(size_t)n * DD + k];
}

// ---------------------------------------------------------------------------
// Pack x [TT*BB, DD] fp32 into f16 A-matrix tiles (ISA 7.12.2, 16-bit A 16x32):
//   lane l holds row m = l%16;
//   elems 0..7  -> K = 8*(l/16) + e ; elems 8..15 -> K = 16 + 8*(l/16) + (e-8)
// Packed index: o = ((mtile*(DD/32) + ktile)*32 + lane)*16 + e
// ---------------------------------------------------------------------------
__global__ __launch_bounds__(256)
void pack_x_f16(const float* __restrict__ X, _Float16* __restrict__ Xp) {
  size_t o = (size_t)blockIdx.x * blockDim.x + threadIdx.x;
  if (o >= (size_t)TT * BB * DD) return;
  int e    = (int)(o & 15);
  int lane = (int)((o >> 4) & 31);
  int kt   = (int)((o >> 9) & (DD / 32 - 1));
  int mt   = (int)(o >> 14);
  int m    = lane & 15;
  int half = lane >> 4;
  int k = kt * 32 + 8 * half + (e & 7) + 16 * (e >> 3);
  Xp[o] = (_Float16)X[(size_t)(mt * 16 + m) * DD + k];
}

// Copy h0 -> d_out[0] (fp32) and seed the f16 h double-buffer.
__global__ __launch_bounds__(256)
void init_h0_kernel(const float* __restrict__ h0, float* __restrict__ out0,
                    _Float16* __restrict__ hbuf0) {
  int i = blockIdx.x * 256 + threadIdx.x;
  if (i < BB * DD) {
    float v = h0[i];
    out0[i]  = v;
    hbuf0[i] = (_Float16)v;
  }
}

// ---------------------------------------------------------------------------
// Phase 1: xp = x @ W_x^T + b  (M = 16384, N = K = 1024), all-f16 operands.
// One wave owns a 16x64 C strip: 1 A-tile x 4 B-tiles -> 4 WMMAs per A load.
// Register double-buffered: iteration k+1 loads issue before iteration k's
// WMMAs so the XDL pipe overlaps L2 fetch latency (no wait-loadcnt-0 stall).
// ---------------------------------------------------------------------------
__global__ __launch_bounds__(128)
void xproj_wmma_kernel(const _Float16* __restrict__ Xp,  // packed A tiles
                       const _Float16* __restrict__ Wxp, // packed B tiles
                       const float* __restrict__ bias,   // [DD]
                       float* __restrict__ xp)           // d_out + BB*DD
{
  const int wave = threadIdx.x >> 5;
  const int lane = threadIdx.x & 31;
  const int wid  = blockIdx.x * 4 + wave;   // 0 .. 16383
  const int ng   = wid & 15;                // 16 groups of 4 n-tiles
  const int mt   = wid >> 4;                // 1024 m-tiles
  const int nt0  = ng * 4;
  const int m    = lane & 15;
  const int half = lane >> 4;

  const _Float16* at = Xp + (size_t)mt * (DD / 32) * 512 + lane * 16;
  const _Float16* b0 = Wxp + (size_t)(nt0 + 0) * (DD / 32) * 512 + lane * 16;
  const _Float16* b1 = Wxp + (size_t)(nt0 + 1) * (DD / 32) * 512 + lane * 16;
  const _Float16* b2 = Wxp + (size_t)(nt0 + 2) * (DD / 32) * 512 + lane * 16;
  const _Float16* b3 = Wxp + (size_t)(nt0 + 3) * (DD / 32) * 512 + lane * 16;

  v8f acc0 = {}, acc1 = {}, acc2 = {}, acc3 = {};

  // Prologue: stage k-step 0
  v16h a   = *(const v16h*)(at);
  v16h vb0 = *(const v16h*)(b0);
  v16h vb1 = *(const v16h*)(b1);
  v16h vb2 = *(const v16h*)(b2);
  v16h vb3 = *(const v16h*)(b3);

#pragma unroll 2
  for (int kt = 0; kt < DD / 32 - 1; ++kt) {
    const size_t ko = (size_t)(kt + 1) * 512;
    // Issue next k-step's loads first ...
    v16h an = *(const v16h*)(at + ko);
    v16h n0 = *(const v16h*)(b0 + ko);
    v16h n1 = *(const v16h*)(b1 + ko);
    v16h n2 = *(const v16h*)(b2 + ko);
    v16h n3 = *(const v16h*)(b3 + ko);
    __builtin_prefetch(at + ko + 512, 0, 1);
    // ... then do this k-step's matrix math while they fly
    acc0 = __builtin_amdgcn_wmma_f32_16x16x32_f16(false, a, false, vb0, (short)0, acc0, false, false);
    acc1 = __builtin_amdgcn_wmma_f32_16x16x32_f16(false, a, false, vb1, (short)0, acc1, false, false);
    acc2 = __builtin_amdgcn_wmma_f32_16x16x32_f16(false, a, false, vb2, (short)0, acc2, false, false);
    acc3 = __builtin_amdgcn_wmma_f32_16x16x32_f16(false, a, false, vb3, (short)0, acc3, false, false);
    a = an; vb0 = n0; vb1 = n1; vb2 = n2; vb3 = n3;
  }
  // Epilogue k-step
  acc0 = __builtin_amdgcn_wmma_f32_16x16x32_f16(false, a, false, vb0, (short)0, acc0, false, false);
  acc1 = __builtin_amdgcn_wmma_f32_16x16x32_f16(false, a, false, vb1, (short)0, acc1, false, false);
  acc2 = __builtin_amdgcn_wmma_f32_16x16x32_f16(false, a, false, vb2, (short)0, acc2, false, false);
  acc3 = __builtin_amdgcn_wmma_f32_16x16x32_f16(false, a, false, vb3, (short)0, acc3, false, false);

  // C/D layout: lane l -> N = l%16, VGPR v -> M = v + 8*(l/16)
  v8f accs[4] = {acc0, acc1, acc2, acc3};
#pragma unroll
  for (int j = 0; j < 4; ++j) {
    const int col = (nt0 + j) * 16 + m;
    const float bv = bias[col];
#pragma unroll
    for (int v = 0; v < 8; ++v) {
      int r = mt * 16 + 8 * half + v;
      xp[(size_t)r * DD + col] = accs[j][v] + bv;
    }
  }
}

// ---------------------------------------------------------------------------
// Phase 2 (one launch per step): h_t = tanh(xp_t + h_{t-1} @ W_h^T)
// 128 C tiles; one block (4 waves) per tile, K split 4 ways (8 WMMAs each,
// register double-buffered), partial sums reduced through LDS. Reads xp in
// place from d_out, writes h_t fp32 back plus an f16 copy for step t+1.
// ---------------------------------------------------------------------------
__global__ __launch_bounds__(128)
void elman_step_kernel(const _Float16* __restrict__ Whp,   // packed B tiles
                       const _Float16* __restrict__ hprev, // [BB, DD] f16
                       float* __restrict__ out_t,          // d_out + (t+1)*BB*DD
                       _Float16* __restrict__ hnext)       // [BB, DD] f16
{
  __shared__ v8f red[3][32];

  const int wave = threadIdx.x >> 5;
  const int lane = threadIdx.x & 31;
  const int tile = blockIdx.x;   // 0 .. 127
  const int nt   = tile & 63;
  const int mt   = tile >> 6;
  const int m    = lane & 15;
  const int half = lane >> 4;

  const _Float16* arow  = hprev + (size_t)(mt * 16 + m) * DD;
  const _Float16* btile = Whp + (size_t)nt * (DD / 32) * 512 + lane * 16;
  const int kt0 = wave * 8;                // this wave's K slice

  v8f acc = {};

  // Prologue: stage first k-step of this wave's slice
  v8h alo = *(const v8h*)(arow + kt0 * 32 + 8 * half);
  v8h ahi = *(const v8h*)(arow + kt0 * 32 + 16 + 8 * half);
  v16h b  = *(const v16h*)(btile + (size_t)kt0 * 512);

#pragma unroll
  for (int i = 0; i < 8; ++i) {
    v8h nlo, nhi; v16h nb;
    if (i < 7) {                           // issue next k-step's loads first
      const int kn = kt0 + i + 1;
      nlo = *(const v8h*)(arow + kn * 32 + 8 * half);
      nhi = *(const v8h*)(arow + kn * 32 + 16 + 8 * half);
      nb  = *(const v16h*)(btile + (size_t)kn * 512);
      __builtin_prefetch(btile + (size_t)(kn + 1) * 512, 0, 1);
    }
    v16h a;
#pragma unroll
    for (int q = 0; q < 8; ++q) { a[q] = alo[q]; a[q + 8] = ahi[q]; }
    acc = __builtin_amdgcn_wmma_f32_16x16x32_f16(false, a, false, b,
                                                 (short)0, acc, false, false);
    if (i < 7) { alo = nlo; ahi = nhi; b = nb; }
  }

  // Cross-wave K reduction through LDS (ds_store/ds_load b128 path)
  if (wave != 0) red[wave - 1][lane] = acc;
  __syncthreads();

  if (wave == 0) {
    acc = acc + red[0][lane] + red[1][lane] + red[2][lane];
    const int col = nt * 16 + m;
#pragma unroll
    for (int v = 0; v < 8; ++v) {
      int r = mt * 16 + 8 * half + v;      // batch row
      size_t idx = (size_t)r * DD + col;
      float pre = out_t[idx] + acc[v];     // xp already there (phase 1)
      float hv  = tanhf(pre);
      out_t[idx] = hv;                     // fp32 result
      hnext[idx] = (_Float16)hv;           // f16 A operand for next step
    }
  }
}

// ---------------------------------------------------------------------------
extern "C" void kernel_launch(void* const* d_in, const int* in_sizes, int n_in,
                              void* d_out, int out_size, void* d_ws, size_t ws_size,
                              hipStream_t stream) {
  const float* x    = (const float*)d_in[0];  // [TT, BB, DD]
  const float* h0   = (const float*)d_in[1];  // [BB, DD]
  const float* Wx   = (const float*)d_in[2];  // [DD, DD]
  const float* Wh   = (const float*)d_in[3];  // [DD, DD]
  const float* bias = (const float*)d_in[4];  // [DD]
  float* out = (float*)d_out;                 // [TT+1, BB, DD]

  char* ws = (char*)d_ws;
  _Float16* Wxp   = (_Float16*)ws;                               // 2 MB
  _Float16* Whp   = (_Float16*)(ws + (size_t)DD * DD * 2);       // 2 MB
  _Float16* hbuf0 = (_Float16*)(ws + (size_t)DD * DD * 4);       // 64 KB
  _Float16* hbuf1 = hbuf0 + (size_t)BB * DD;                     // 64 KB
  _Float16* Xp    = (_Float16*)(ws + (size_t)DD * DD * 4
                                   + (size_t)2 * BB * DD * 2);   // 32 MB

  // One-time packs (bandwidth-trivial vs 23.3 TB/s)
  pack_weight_f16<<<(DD * DD) / 256, 256, 0, stream>>>(Wx, Wxp);
  pack_weight_f16<<<(DD * DD) / 256, 256, 0, stream>>>(Wh, Whp);
  pack_x_f16<<<(TT * BB * DD) / 256, 256, 0, stream>>>(x, Xp);
  init_h0_kernel<<<(BB * DD + 255) / 256, 256, 0, stream>>>(h0, out, hbuf0);

  // Phase 1: 1024 m-tiles x 16 n-groups = 16384 waves, 4 waves/block
  xproj_wmma_kernel<<<16384 / 4, 128, 0, stream>>>(
      Xp, Wxp, bias, out + (size_t)BB * DD);

  // Phase 2: 512 dependent steps, stream order = global sync
  for (int t = 0; t < TT; ++t) {
    _Float16* hp = (t & 1) ? hbuf1 : hbuf0;
    _Float16* hn = (t & 1) ? hbuf0 : hbuf1;
    elman_step_kernel<<<128, 128, 0, stream>>>(
        Whp, hp, out + (size_t)(t + 1) * BB * DD, hn);
  }
}
